// CausalSelfAttentionModern_5368709120031
// MI455X (gfx1250) — matrix operand order repaired and checked by reading, not compiled, not run on hardware
//
#include <hip/hip_runtime.h>
#include <stdint.h>

// ---------------------------------------------------------------------------
// CDNA5 (gfx1250) causal GQA attention block, bf16 WMMA + f32 accumulate.
// All WMMA fragment loads are contiguous 16/32-byte vector loads; K/V and
// GEMM tiles are staged with async global->LDS copies, double-buffered.
// ---------------------------------------------------------------------------

typedef __attribute__((ext_vector_type(16))) __bf16 v16bf;
typedef __attribute__((ext_vector_type(8)))  __bf16 v8bf;
typedef __attribute__((ext_vector_type(8)))  float  v8f;
typedef int v4i __attribute__((__vector_size__(4 * sizeof(int))));

union bfu { v16bf v; v8bf h[2]; };

#define AS1 __attribute__((address_space(1)))
#define AS3 __attribute__((address_space(3)))

#if defined(__HIP_DEVICE_COMPILE__) && __has_builtin(__builtin_amdgcn_global_load_async_to_lds_b128)
#define HAS_ASYNC_LDS 1
#endif

__device__ __forceinline__ void copy16_to_lds(const __bf16* g, __bf16* l) {
#if defined(HAS_ASYNC_LDS)
  __builtin_amdgcn_global_load_async_to_lds_b128((AS1 v4i*)(void*)g,
                                                 (AS3 v4i*)(void*)l,
                                                 0, 0);
#else
  *(uint4*)l = *(const uint4*)g;
#endif
}

__device__ __forceinline__ void wait_async_lds() {
#if defined(HAS_ASYNC_LDS)
#if __has_builtin(__builtin_amdgcn_s_wait_asynccnt)
  __builtin_amdgcn_s_wait_asynccnt(0);
#else
  asm volatile("s_wait_asynccnt 0" ::: "memory");
#endif
#endif
}

// WMMA fragment maps as used here (wave32; A/B use contiguous k, not the ISA table order):
//  A 16x32 bf16 : lane m=lane&15 holds row m; elem e (hi=lane>>4):
//                 K = e + 16*hi  -> one contiguous 16-run at
//                 K0 = 16*hi (same k order as B)
//  B 32x16 bf16 : lane n=lane&15 holds col n; elem e: K = e + 16*hi
//                 -> one contiguous 16-run at K0 = 16*hi
//  C 16x16 f32  : vgpr r, lane: (M = r + 8*hi, N = lane&15)

__device__ __forceinline__ v16bf load_a_frag(const __bf16* rowbase, int hi) {
  return *(const v16bf*)(rowbase + 16 * hi);
}

__device__ __forceinline__ v16bf load_b_frag(const __bf16* colbase, int hi) {
  return *(const v16bf*)(colbase + 16 * hi);
}

constexpr int T_SEQ = 2048;
constexpr int C_EMB = 2048;
constexpr int NH    = 16;
constexpr int NKV   = 4;
constexpr int DH    = 128;
constexpr int CQKV  = NH * DH + 2 * NKV * DH; // 3072

// ---------------------------------------------------------------------------
// Weight packing (transposed to K-major so GEMM B tiles stage contiguously)
// ---------------------------------------------------------------------------
__global__ __launch_bounds__(256) void cvt_bf16_kernel(const float* __restrict__ s,
                                                       __bf16* __restrict__ d, int n) {
  for (int i = blockIdx.x * blockDim.x + threadIdx.x; i < n; i += gridDim.x * blockDim.x)
    d[i] = (__bf16)s[i];
}

// dst[c*rows + r] = src[r*cols + c]  (bf16 transpose-convert)
__global__ __launch_bounds__(256) void cvt_bf16_T_kernel(const float* __restrict__ s,
                                                         __bf16* __restrict__ d,
                                                         int rows, int cols) {
  int n = rows * cols;
  for (int i = blockIdx.x * blockDim.x + threadIdx.x; i < n; i += gridDim.x * blockDim.x) {
    int c = i / rows, r = i - c * rows;
    d[i] = (__bf16)s[(size_t)r * cols + c];
  }
}

// Packed, transposed QKV weight: wpT[c*2048 + r], c in [0,3072)
__global__ __launch_bounds__(256) void pack_qkvw_T_kernel(const float* __restrict__ wq,
                                                          const float* __restrict__ wk,
                                                          const float* __restrict__ wv,
                                                          __bf16* __restrict__ wpT) {
  const int n = C_EMB * CQKV;
  for (int i = blockIdx.x * blockDim.x + threadIdx.x; i < n; i += gridDim.x * blockDim.x) {
    int c = i >> 11, r = i & (C_EMB - 1);
    float v;
    if (c < NH * DH)                 v = wq[(size_t)r * (NH * DH) + c];
    else if (c < NH * DH + NKV * DH) v = wk[(size_t)r * (NKV * DH) + (c - NH * DH)];
    else                             v = wv[(size_t)r * (NKV * DH) + (c - NH * DH - NKV * DH)];
    wpT[i] = (__bf16)v;
  }
}

// ---------------------------------------------------------------------------
// Tiled bf16 GEMM: C(f32, MxN) = A(bf16 row-major MxK) * Bt(bf16 K-major NxK)
// Block: 256 threads (8 waves). Tile 128x128. K step 32, double-buffered
// async LDS staging; B-frags preloaded so 8 WMMAs issue back-to-back.
// ---------------------------------------------------------------------------
__global__ __launch_bounds__(256) void gemm_bf16_kernel(const __bf16* __restrict__ A,
                                                        const __bf16* __restrict__ Bt,
                                                        float* __restrict__ Cout,
                                                        int M, int N, int K) {
  __shared__ __attribute__((aligned(64))) __bf16 sA[2][128 * 32];   // [row][K]
  __shared__ __attribute__((aligned(64))) __bf16 sBt[2][128 * 32];  // [col][K]

  const int tid  = threadIdx.x;
  const int wav  = tid >> 5;
  const int lane = tid & 31;
  const int n16  = lane & 15;
  const int hi   = lane >> 4;
  const int mBase = blockIdx.y * 128;
  const int nBase = blockIdx.x * 128;

  v8f acc[8];
#pragma unroll
  for (int i = 0; i < 8; ++i) acc[i] = (v8f)(0.0f);

  auto stage = [&](int buf, int k0) {
#pragma unroll
    for (int i = 0; i < 2; ++i) {
      int c = tid + 256 * i;                 // 512 16B chunks per matrix tile
      int rr = c >> 2, ko = (c & 3) * 8;     // 128 rows x 32 K each
      copy16_to_lds(&A[(size_t)(mBase + rr) * K + k0 + ko], &sA[buf][rr * 32 + ko]);
      copy16_to_lds(&Bt[(size_t)(nBase + rr) * K + k0 + ko], &sBt[buf][rr * 32 + ko]);
    }
  };

  const int nk = K / 32;
  stage(0, 0);
  for (int kt = 0; kt < nk; ++kt) {
    wait_async_lds();
    __syncthreads();
    if (kt + 1 < nk) stage((kt + 1) & 1, (kt + 1) * 32);

    const __bf16* a_ = &sA[kt & 1][0];
    const __bf16* b_ = &sBt[kt & 1][0];

    v16bf av = load_a_frag(a_ + (wav * 16 + n16) * 32, hi);
    v16bf bv[8];
#pragma unroll
    for (int nb = 0; nb < 8; ++nb)
      bv[nb] = load_b_frag(b_ + (nb * 16 + n16) * 32, hi);
#pragma unroll
    for (int nb = 0; nb < 8; ++nb)
      acc[nb] = __builtin_amdgcn_wmma_f32_16x16x32_bf16(
          false, av, false, bv[nb], (short)0, acc[nb], false, false);
    __syncthreads();
  }

#pragma unroll
  for (int nb = 0; nb < 8; ++nb)
#pragma unroll
    for (int r = 0; r < 8; ++r) {
      int row = mBase + wav * 16 + r + 8 * hi;
      int col = nBase + nb * 16 + n16;
      Cout[(size_t)row * N + col] = acc[nb][r];
    }
}

// ---------------------------------------------------------------------------
// RoPE + head-major bf16 split of fused QKV (f32 in, bf16 out)
// qkv: [T, 3072] f32 -> Qb[NH][T][DH], Kb[NKV][T][DH] (RoPE'd),
//                       Vt[NKV][DH][T] (d-major, for contiguous B-frags)
// ---------------------------------------------------------------------------
__global__ __launch_bounds__(256) void rope_split_kernel(const float* __restrict__ qkv,
                                                         __bf16* __restrict__ Qb,
                                                         __bf16* __restrict__ Kb,
                                                         __bf16* __restrict__ Vt) {
  const int t   = blockIdx.x;
  const int tid = threadIdx.x;
  const float* row = qkv + (size_t)t * CQKV;

  for (int idx = tid; idx < NH * DH; idx += 256) {
    int h = idx >> 7, d = idx & 127, j = d & 63;
    float inv = __powf(10000.0f, -(float)j * (1.0f / 64.0f));
    float sn, cs;
    __sincosf((float)t * inv, &sn, &cs);
    float x0 = row[idx];
    float xr = (d < 64) ? -row[h * DH + d + 64] : row[h * DH + d - 64];
    Qb[((size_t)h * T_SEQ + t) * DH + d] = (__bf16)(x0 * cs + xr * sn);
  }
  for (int idx = tid; idx < NKV * DH; idx += 256) {
    int h = idx >> 7, d = idx & 127, j = d & 63;
    float inv = __powf(10000.0f, -(float)j * (1.0f / 64.0f));
    float sn, cs;
    __sincosf((float)t * inv, &sn, &cs);
    float x0 = row[NH * DH + idx];
    float xr = (d < 64) ? -row[NH * DH + h * DH + d + 64] : row[NH * DH + h * DH + d - 64];
    Kb[((size_t)h * T_SEQ + t) * DH + d] = (__bf16)(x0 * cs + xr * sn);
  }
  for (int idx = tid; idx < NKV * DH; idx += 256) {
    int h = idx >> 7, d = idx & 127;
    Vt[((size_t)h * DH + d) * T_SEQ + t] = (__bf16)row[NH * DH + NKV * DH + idx];
  }
}

// ---------------------------------------------------------------------------
// Flash-style causal GQA attention. Block = 4 waves (128 thr).
// grid: (head, qtile); each wave owns 16 query rows, WG owns 64.
// Per 32-key block: S = Q K^T (8 WMMA), online softmax (f32), O += P V (8 WMMA).
// K/V tiles double-buffered via async global->LDS copies.
// ---------------------------------------------------------------------------
__global__ __launch_bounds__(128) void attn_kernel(const __bf16* __restrict__ Q,
                                                   const __bf16* __restrict__ K,
                                                   const __bf16* __restrict__ Vt,
                                                   __bf16* __restrict__ Yout) {
  __shared__ __attribute__((aligned(64))) __bf16 ldsK[2][32 * DH];  // [key][d]
  __shared__ __attribute__((aligned(64))) __bf16 ldsVt[2][DH * 32]; // [d][key]
  __shared__ __attribute__((aligned(64))) __bf16 ldsP[4][16 * 32];  // [row][key]

  const int head = blockIdx.x;
  const int qt   = blockIdx.y;
  const int kvh  = head / (NH / NKV);
  const int tid  = threadIdx.x;
  const int wav  = tid >> 5;
  const int lane = tid & 31;
  const int n16  = lane & 15;
  const int hi   = lane >> 4;

  const int qbase = qt * 64;
  const int row0  = qbase + wav * 16;

  // Q fragments: 16 rows x 128 dims = 4 A-frags of 16x32 (contiguous loads)
  v16bf aq[4];
  const __bf16* Qh = Q + ((size_t)head * T_SEQ + row0) * DH;
#pragma unroll
  for (int c = 0; c < 4; ++c)
    aq[c] = load_a_frag(Qh + (size_t)n16 * DH + 32 * c, hi);

  v8f o[8];
#pragma unroll
  for (int i = 0; i < 8; ++i) o[i] = (v8f)(0.0f);
  float m_r[8], l_r[8];
#pragma unroll
  for (int r = 0; r < 8; ++r) { m_r[r] = -3.0e38f; l_r[r] = 0.0f; }

  const __bf16* Kh  = K  + (size_t)kvh * T_SEQ * DH;
  const __bf16* Vth = Vt + (size_t)kvh * DH * T_SEQ;
  const float scale = 0.08838834764831845f; // 1/sqrt(128)

  auto stage = [&](int buf, int kb) {
#pragma unroll
    for (int i = 0; i < 4; ++i) {
      int c = tid + 128 * i;                 // 512 chunks per matrix
      int kr = c >> 4, kc = (c & 15) * 8;    // K: 32 rows x 128 d
      copy16_to_lds(&Kh[(size_t)(kb * 32 + kr) * DH + kc], &ldsK[buf][kr * DH + kc]);
      int vr = c >> 2, vc = (c & 3) * 8;     // Vt: 128 rows x 32 t
      copy16_to_lds(&Vth[(size_t)vr * T_SEQ + kb * 32 + vc], &ldsVt[buf][vr * 32 + vc]);
    }
  };

  const int nkb = (qbase + 64) / 32;
  stage(0, 0);
  for (int kb = 0; kb < nkb; ++kb) {
    wait_async_lds();
    __syncthreads();
    if (kb + 1 < nkb) stage((kb + 1) & 1, kb + 1);

    const __bf16* kbuf = &ldsK[kb & 1][0];
    const __bf16* vbuf = &ldsVt[kb & 1][0];

    // S tiles: 16 q-rows x 32 keys as two 16x16 f32 C frags.
    // Preload all 8 B-frags (contiguous in d with K stored [key][d]),
    // then issue the 8 WMMAs back-to-back.
    v16bf bs[8];
#pragma unroll
    for (int c = 0; c < 4; ++c) {
      bs[2 * c]     = *(const v16bf*)&kbuf[(size_t)n16 * DH + 32 * c + 16 * hi];
      bs[2 * c + 1] = *(const v16bf*)&kbuf[(size_t)(16 + n16) * DH + 32 * c + 16 * hi];
    }
    v8f s0 = (v8f)(0.0f), s1 = (v8f)(0.0f);
#pragma unroll
    for (int c = 0; c < 4; ++c) {
      s0 = __builtin_amdgcn_wmma_f32_16x16x32_bf16(false, aq[c], false, bs[2 * c],     (short)0, s0, false, false);
      s1 = __builtin_amdgcn_wmma_f32_16x16x32_bf16(false, aq[c], false, bs[2 * c + 1], (short)0, s1, false, false);
    }

    // online softmax, causal mask folded in
    const int kg0 = kb * 32;
#pragma unroll
    for (int r = 0; r < 8; ++r) {
      int qrow = row0 + r + 8 * hi;
      float v0 = (kg0 + n16      <= qrow) ? s0[r] * scale : -3.0e38f;
      float v1 = (kg0 + 16 + n16 <= qrow) ? s1[r] * scale : -3.0e38f;
      float mx = fmaxf(v0, v1);
#pragma unroll
      for (int sh = 1; sh < 16; sh <<= 1) mx = fmaxf(mx, __shfl_xor(mx, sh, 32));
      float mnew  = fmaxf(m_r[r], mx);
      float alpha = __expf(m_r[r] - mnew);
      float e0 = __expf(v0 - mnew);
      float e1 = __expf(v1 - mnew);
      float rs = e0 + e1;
#pragma unroll
      for (int sh = 1; sh < 16; sh <<= 1) rs += __shfl_xor(rs, sh, 32);
      l_r[r] = l_r[r] * alpha + rs;
      m_r[r] = mnew;
#pragma unroll
      for (int nb = 0; nb < 8; ++nb) o[nb][r] *= alpha;
      // P (bf16) to per-wave LDS scratch, row-major 16x32
      ldsP[wav][(r + 8 * hi) * 32 + n16]      = (__bf16)e0;
      ldsP[wav][(r + 8 * hi) * 32 + 16 + n16] = (__bf16)e1;
    }
    asm volatile("" ::: "memory"); // keep P stores before P loads

    // O += P * V : preload 8 V B-frags (contiguous in key), then 8 WMMAs.
    v16bf pa = load_a_frag(&ldsP[wav][n16 * 32], hi);
    v16bf bv[8];
#pragma unroll
    for (int nb = 0; nb < 8; ++nb)
      bv[nb] = load_b_frag(&vbuf[(nb * 16 + n16) * 32], hi);
#pragma unroll
    for (int nb = 0; nb < 8; ++nb)
      o[nb] = __builtin_amdgcn_wmma_f32_16x16x32_bf16(false, pa, false, bv[nb], (short)0, o[nb], false, false);
    __syncthreads();
  }

  // epilogue: O / l, write bf16 attn output row-major [T, 2048]
#pragma unroll
  for (int nb = 0; nb < 8; ++nb)
#pragma unroll
    for (int r = 0; r < 8; ++r) {
      int row = row0 + r + 8 * hi;
      int col = head * DH + nb * 16 + n16;
      Yout[(size_t)row * C_EMB + col] = (__bf16)(o[nb][r] / l_r[r]);
    }
}

// ---------------------------------------------------------------------------
// Host launch
// ---------------------------------------------------------------------------
extern "C" void kernel_launch(void* const* d_in, const int* in_sizes, int n_in,
                              void* d_out, int out_size, void* d_ws, size_t ws_size,
                              hipStream_t stream) {
  const float* x     = (const float*)d_in[0];
  const float* wq    = (const float*)d_in[1];
  const float* wk    = (const float*)d_in[2];
  const float* wv    = (const float*)d_in[3];
  const float* wproj = (const float*)d_in[4];
  float* out = (float*)d_out;

  char* w = (char*)d_ws;
  size_t off = 0;
  auto alloc = [&](size_t bytes) {
    void* p = w + off;
    off += (bytes + 255) & ~(size_t)255;
    return p;
  };
  __bf16* xbf      = (__bf16*)alloc((size_t)T_SEQ * C_EMB * 2);
  __bf16* wpackT   = (__bf16*)alloc((size_t)C_EMB * CQKV * 2);
  __bf16* wprojTbf = (__bf16*)alloc((size_t)C_EMB * C_EMB * 2);
  float*  qkv      = (float*) alloc((size_t)T_SEQ * CQKV * 4);
  __bf16* Qb       = (__bf16*)alloc((size_t)NH  * T_SEQ * DH * 2);
  __bf16* Kb       = (__bf16*)alloc((size_t)NKV * T_SEQ * DH * 2);
  __bf16* Vtb      = (__bf16*)alloc((size_t)NKV * DH * T_SEQ * 2);
  __bf16* attnbf   = (__bf16*)alloc((size_t)T_SEQ * C_EMB * 2);

  cvt_bf16_kernel<<<4096, 256, 0, stream>>>(x, xbf, T_SEQ * C_EMB);
  pack_qkvw_T_kernel<<<4096, 256, 0, stream>>>(wq, wk, wv, wpackT);
  cvt_bf16_T_kernel<<<4096, 256, 0, stream>>>(wproj, wprojTbf, C_EMB, C_EMB);

  // fused QKV projection: [2048x2048] x [2048x3072] -> f32
  gemm_bf16_kernel<<<dim3(CQKV / 128, T_SEQ / 128), 256, 0, stream>>>(
      xbf, wpackT, qkv, T_SEQ, CQKV, C_EMB);

  rope_split_kernel<<<T_SEQ, 256, 0, stream>>>(qkv, Qb, Kb, Vtb);

  attn_kernel<<<dim3(NH, T_SEQ / 64), 128, 0, stream>>>(Qb, Kb, Vtb, attnbf);

  // output projection -> f32 d_out
  gemm_bf16_kernel<<<dim3(C_EMB / 128, T_SEQ / 128), 256, 0, stream>>>(
      attnbf, wprojTbf, out, T_SEQ, C_EMB, C_EMB);
}
